// MultiHeadSelfAttention_23888608101368
// MI455X (gfx1250) — matrix operand order repaired and checked by reading, not compiled, not run on hardware
//
#include <hip/hip_runtime.h>

#define DMODEL 1024
#define NHEAD  16
#define HDIM   64
#define NTOK   2048
#define NBATCH 2
#define MROWS  (NBATCH * NTOK)   // 4096
#define NQKV   (3 * DMODEL)      // 3072

// LDS strides (in halves). Multiples of 8 halves (16B) so B128 LDS ops are
// naturally aligned; 40 halves = 20 dwords and 72 halves = 36 dwords both give
// 16 disjoint 4-bank groups across lanes 0-15 -> conflict-free ds_load_b128.
#define XS_STRIDE 40
#define WS_STRIDE 40
#define KS_STRIDE 72
#define VS_STRIDE 40
#define PS_STRIDE 40

typedef __attribute__((ext_vector_type(16))) _Float16 v16h;
typedef __attribute__((ext_vector_type(8)))  float    v8f;

union Frag { v16h h; unsigned u[8]; uint4 q[2]; };
union H8   { uint4 u4; _Float16 h[8]; };
union H4   { uint2 u2; _Float16 h[4]; };

__device__ __forceinline__ v8f wmma_f32(v16h a, v16h b, v8f c) {
  // (neg_a, A, neg_b, B, c_mod, C, reuse_a, reuse_b)
  return __builtin_amdgcn_wmma_f32_16x16x32_f16(false, a, false, b, (short)0, c, false, false);
}

// Async 16B global->LDS copy (ASYNCcnt). LDS operand: low 32 bits of the
// generic pointer are the LDS byte address (flat aperture rule).
__device__ __forceinline__ void async_b128(const _Float16* g, _Float16* l) {
  asm volatile("global_load_async_to_lds_b128 %0, %1, off"
               :: "v"((unsigned)(uintptr_t)l), "v"(g) : "memory");
}
__device__ __forceinline__ void wait_async0() {
  asm volatile("s_wait_asynccnt 0x0" ::: "memory");
}

// A fragment (16x32 f16, MxK) from LDS row-major [m][k].
// Code's order: lane m=l&15; elem e -> k=(l>>4)*16+e (contiguous, matches B). One 32B k-contiguous run
// -> two ds_load_b128 (stride must be mult. of 8).
__device__ __forceinline__ v16h load_fragA(const _Float16* base, int stride, int lane) {
  Frag f;
  const int m = lane & 15, hh = lane >> 4;
  const _Float16* row = base + m * stride + (hh << 4);
  f.q[0] = *(const uint4*)(row);
  f.q[1] = *(const uint4*)(row + 8);
  return f.h;
}

// B fragment (32x16 f16, KxN) from LDS laid out [n][k].
// Code's order: lane n=l&15; elem e -> k=(l>>4)*16+e (contiguous, matches A). One 32B k-contiguous run.
__device__ __forceinline__ v16h load_fragB(const _Float16* base, int stride, int lane) {
  Frag f;
  const int n = lane & 15, hh = lane >> 4;
  const _Float16* row = base + n * stride + (hh << 4);
  f.q[0] = *(const uint4*)(row);
  f.q[1] = *(const uint4*)(row + 8);
  return f.h;
}

// Reductions over 16-lane halves (C/D rows live in one half; xor masks < 16
// never cross the wave32 half boundary).
__device__ __forceinline__ float redmax16(float x) {
#pragma unroll
  for (int m = 1; m < 16; m <<= 1) x = fmaxf(x, __shfl_xor(x, m, 32));
  return x;
}
__device__ __forceinline__ float redsum16(float x) {
#pragma unroll
  for (int m = 1; m < 16; m <<= 1) x += __shfl_xor(x, m, 32);
  return x;
}

__global__ void cvt_f32_to_f16(const float4* __restrict__ src, _Float16* __restrict__ dst, int n4) {
  const int i = blockIdx.x * 256 + threadIdx.x;
  if (i < n4) {
    const float4 f = src[i];
    H4 o;
    o.h[0] = (_Float16)f.x; o.h[1] = (_Float16)f.y;
    o.h[2] = (_Float16)f.z; o.h[3] = (_Float16)f.w;
    *(uint2*)(dst + 4 * (size_t)i) = o.u2;
  }
}

// ---- shared WMMA GEMM mainloop: 128x64 block tile, K=1024 in steps of 32,
// double-buffered so the async global->LDS copy of tile i+1 overlaps the
// WMMAs of tile i. 8 waves as 4(M) x 2(N); each wave: 32x32 = 2x2 frags.
__device__ __forceinline__ void gemm_mainloop(const _Float16* __restrict__ X,
                                              const _Float16* __restrict__ W,
                                              int ldw, int rowBase, int colBase,
                                              _Float16* Xs, _Float16* Ws,  // 2x buffers each
                                              v8f (&acc)[2][2]) {
  const int tid = threadIdx.x, lane = tid & 31, wid = tid >> 5;
  const int waveM = wid >> 1, waveN = wid & 1;

  // X tile 128x32 halves = 512 x B128 chunks; this thread owns 2 (rows r, r+64).
  const int xr = tid >> 2, xc = (tid & 3) << 3;
  const _Float16* xg = X + (size_t)(rowBase + xr) * DMODEL + xc;
  const int xoff = xr * XS_STRIDE + xc;
  // W tile 32x64: this thread owns one B128 (k row, 8 n) -> transpose-scatter.
  const int wk = tid >> 3, wn = (tid & 7) << 3;
  const _Float16* wg = W + (size_t)wk * ldw + colBase + wn;

  auto stage = [&](int k0, int buf) {
    _Float16* xb = Xs + buf * (128 * XS_STRIDE) + xoff;
    async_b128(xg + k0, xb);
    async_b128(xg + k0 + (size_t)64 * DMODEL, xb + 64 * XS_STRIDE);
    H8 d;
    d.u4 = *(const uint4*)(wg + (size_t)k0 * ldw);
    _Float16* wb = Ws + buf * (64 * WS_STRIDE);
#pragma unroll
    for (int e = 0; e < 8; ++e) wb[(wn + e) * WS_STRIDE + wk] = d.h[e];
  };

  stage(0, 0);  // prologue
  for (int it = 0; it < DMODEL / 32; ++it) {
    wait_async0();     // my async copy of tile `it` done
    __syncthreads();   // all waves staged tile `it`; all reads of tile it-1 done
    const int cur = it & 1;
    if (it + 1 < DMODEL / 32) {
      stage((it + 1) * 32, cur ^ 1);               // overlaps with WMMAs below
      __builtin_prefetch(xg + (it + 2) * 32, 0, 1);  // warm L2 for tile after next
    }
    const _Float16* xb = Xs + cur * (128 * XS_STRIDE);
    const _Float16* wb = Ws + cur * (64 * WS_STRIDE);
    const v16h a0 = load_fragA(xb + (waveM * 32) * XS_STRIDE,      XS_STRIDE, lane);
    const v16h a1 = load_fragA(xb + (waveM * 32 + 16) * XS_STRIDE, XS_STRIDE, lane);
    const v16h b0 = load_fragB(wb + (waveN * 32) * WS_STRIDE,      WS_STRIDE, lane);
    const v16h b1 = load_fragB(wb + (waveN * 32 + 16) * WS_STRIDE, WS_STRIDE, lane);
    acc[0][0] = wmma_f32(a0, b0, acc[0][0]);
    acc[0][1] = wmma_f32(a0, b1, acc[0][1]);
    acc[1][0] = wmma_f32(a1, b0, acc[1][0]);
    acc[1][1] = wmma_f32(a1, b1, acc[1][1]);
  }
}

// ---- GEMM 1: qkv = x @ Wqkv + bqkv; scatter to q/k/v [B*H][N][64] f16, q pre-scaled ----
__global__ __launch_bounds__(256) void gemm_qkv_kernel(const _Float16* __restrict__ X,
                                                       const _Float16* __restrict__ W,
                                                       const float* __restrict__ bias,
                                                       _Float16* __restrict__ qb,
                                                       _Float16* __restrict__ kb,
                                                       _Float16* __restrict__ vb) {
  __shared__ _Float16 Xs[2 * 128 * XS_STRIDE];
  __shared__ _Float16 Ws[2 * 64 * WS_STRIDE];
  const int rowBase = blockIdx.y * 128, colBase = blockIdx.x * 64;
  v8f acc[2][2] = {};
  gemm_mainloop(X, W, NQKV, rowBase, colBase, Xs, Ws, acc);
  const int lane = threadIdx.x & 31, wid = threadIdx.x >> 5;
  const int waveM = wid >> 1, waveN = wid & 1;
  const int n = lane & 15, hh = lane >> 4;
#pragma unroll
  for (int i = 0; i < 2; ++i)
#pragma unroll
    for (int j = 0; j < 2; ++j) {
      const int col = colBase + waveN * 32 + j * 16 + n;     // C/D: n = lane&15
      const int which = col >> 10, d = col & 1023;
      const int h = d >> 6, hd = d & 63;
      const float bv = bias[col];
      _Float16* dst = (which == 0) ? qb : ((which == 1) ? kb : vb);
      const float scl = (which == 0) ? 0.125f : 1.0f;        // fold HD^-0.5 into q
#pragma unroll
      for (int v = 0; v < 8; ++v) {
        const int row = rowBase + waveM * 32 + i * 16 + (hh << 3) + v;  // C/D: m=(l>>4)*8+v
        const int b = row >> 11, t = row & 2047;
        dst[((size_t)(b * NHEAD + h) * NTOK + t) * HDIM + hd] = (_Float16)((acc[i][j][v] + bv) * scl);
      }
    }
}

// ---- Flash attention: one block per (b,h, 64-query tile); 4 waves x 16 query rows.
// K/V tiles double-buffered: async copy of key-tile i+1 overlaps tile i's WMMAs.
__global__ __launch_bounds__(128) void attn_kernel(const _Float16* __restrict__ qb,
                                                   const _Float16* __restrict__ kb,
                                                   const _Float16* __restrict__ vb,
                                                   _Float16* __restrict__ attnb) {
  __shared__ _Float16 Ks[2 * 32 * KS_STRIDE];  // [key][hd]  natural: B-frag layout for Q@K^T
  __shared__ _Float16 Vs[2 * 64 * VS_STRIDE];  // [hd][key]  transposed: B-frag layout for P@V
  __shared__ _Float16 Ps[4 * 16 * PS_STRIDE];  // per-wave P relayout C/D -> A
  const int tid = threadIdx.x, lane = tid & 31, wid = tid >> 5;
  const int bh = blockIdx.y;                   // b*16+h
  const int b = bh >> 4, h = bh & 15;
  const int tBase = blockIdx.x * 64;
  const size_t headoff = (size_t)bh * NTOK * HDIM;
  const _Float16* qh = qb + headoff;
  const _Float16* kh = kb + headoff;
  const _Float16* vh = vb + headoff;
  const int mlo = lane & 15, hh = lane >> 4;

  // Q fragments (A-layout) straight from global: two B128 per 32-wide K chunk.
  Frag aq[2];
  {
    const _Float16* qrow = qh + (size_t)(tBase + wid * 16 + mlo) * HDIM + (hh << 4);
#pragma unroll
    for (int c = 0; c < 2; ++c) {
      aq[c].q[0] = *(const uint4*)(qrow + c * 32);
      aq[c].q[1] = *(const uint4*)(qrow + c * 32 + 8);
    }
  }

  // K tile: 32x64 halves = 256 B128 chunks; thread owns keys kr, kr+16.
  const int kr = tid >> 3, kc = (tid & 7) << 3;
  const _Float16* kg = kh + (size_t)kr * HDIM + kc;
  const int koff = kr * KS_STRIDE + kc;

  auto stage = [&](int kt, int buf) {
    const size_t kOff = (size_t)kt * 32 * HDIM;
    _Float16* kl = Ks + buf * (32 * KS_STRIDE) + koff;
    async_b128(kg + kOff, kl);
    async_b128(kg + kOff + 16 * HDIM, kl + 16 * KS_STRIDE);
    _Float16* vl = Vs + buf * (64 * VS_STRIDE);
#pragma unroll
    for (int i = 0; i < 2; ++i) {
      const int key = kr + i * 16;
      H8 d;
      d.u4 = *(const uint4*)(vh + kOff + (size_t)key * HDIM + kc);
#pragma unroll
      for (int e = 0; e < 8; ++e) vl[(kc + e) * VS_STRIDE + key] = d.h[e];
    }
  };

  float m_run[8], l_run[8];
  v8f o[4] = {};
#pragma unroll
  for (int v = 0; v < 8; ++v) { m_run[v] = -1e30f; l_run[v] = 0.0f; }

  stage(0, 0);  // prologue
  for (int kt = 0; kt < NTOK / 32; ++kt) {
    wait_async0();
    __syncthreads();
    const int cur = kt & 1;
    if (kt + 1 < NTOK / 32) stage(kt + 1, cur ^ 1);  // overlaps compute below
    const _Float16* kbuf = Ks + cur * (32 * KS_STRIDE);
    const _Float16* vbuf = Vs + cur * (64 * VS_STRIDE);

    // S(16x32) = Q(16x64) @ K^T : two 16x16 accs over two hd chunks
    v8f s0 = {}, s1 = {};
#pragma unroll
    for (int c = 0; c < 2; ++c) {
      Frag b0f, b1f;
      const _Float16* r0 = kbuf + mlo * KS_STRIDE + c * 32 + (hh << 4);         // keys 0..15
      const _Float16* r1 = kbuf + (mlo + 16) * KS_STRIDE + c * 32 + (hh << 4);  // keys 16..31
      b0f.q[0] = *(const uint4*)(r0); b0f.q[1] = *(const uint4*)(r0 + 8);
      b1f.q[0] = *(const uint4*)(r1); b1f.q[1] = *(const uint4*)(r1 + 8);
      s0 = wmma_f32(aq[c].h, b0f.h, s0);
      s1 = wmma_f32(aq[c].h, b1f.h, s1);
    }

    // online softmax in C/D layout (row = (l>>4)*8+v spans one 16-lane half)
    float corr[8], pA[8], pB[8];
#pragma unroll
    for (int v = 0; v < 8; ++v) {
      const float tm = redmax16(fmaxf(s0[v], s1[v]));
      const float mn = fmaxf(m_run[v], tm);
      corr[v] = __expf(m_run[v] - mn);
      m_run[v] = mn;
      pA[v] = __expf(s0[v] - mn);
      pB[v] = __expf(s1[v] - mn);
      l_run[v] = l_run[v] * corr[v] + redsum16(pA[v] + pB[v]);
    }
#pragma unroll
    for (int j = 0; j < 4; ++j)
#pragma unroll
      for (int v = 0; v < 8; ++v) o[j][v] *= corr[v];

    // relayout P: C/D -> A via per-wave LDS slice (same-wave DS ops are in-order)
    _Float16* Pw = Ps + wid * 16 * PS_STRIDE;
#pragma unroll
    for (int v = 0; v < 8; ++v) {
      const int row = (hh << 3) + v;
      Pw[row * PS_STRIDE + mlo]      = (_Float16)pA[v];
      Pw[row * PS_STRIDE + 16 + mlo] = (_Float16)pB[v];
    }
    asm volatile("" ::: "memory");
    const v16h aP = load_fragA(Pw, PS_STRIDE, lane);
#pragma unroll
    for (int j = 0; j < 4; ++j) {  // O(16x64) += P(16x32) @ V(32x64)
      const v16h bv = load_fragB(vbuf + (j * 16) * VS_STRIDE, VS_STRIDE, lane);
      o[j] = wmma_f32(aP, bv, o[j]);
    }
  }

  // O /= l; write f16 [B][N][H*64+hd] for the output projection
#pragma unroll
  for (int v = 0; v < 8; ++v) l_run[v] = 1.0f / l_run[v];
#pragma unroll
  for (int j = 0; j < 4; ++j)
#pragma unroll
    for (int v = 0; v < 8; ++v) {
      const int t = tBase + wid * 16 + (hh << 3) + v;
      const int hd = j * 16 + mlo;
      attnb[((size_t)b * NTOK + t) * DMODEL + h * HDIM + hd] = (_Float16)(o[j][v] * l_run[v]);
    }
}

// ---- GEMM 2: out = attn @ Wout + bout (f32 result) ----
__global__ __launch_bounds__(256) void gemm_out_kernel(const _Float16* __restrict__ X,
                                                       const _Float16* __restrict__ W,
                                                       const float* __restrict__ bias,
                                                       float* __restrict__ out) {
  __shared__ _Float16 Xs[2 * 128 * XS_STRIDE];
  __shared__ _Float16 Ws[2 * 64 * WS_STRIDE];
  const int rowBase = blockIdx.y * 128, colBase = blockIdx.x * 64;
  v8f acc[2][2] = {};
  gemm_mainloop(X, W, DMODEL, rowBase, colBase, Xs, Ws, acc);
  const int lane = threadIdx.x & 31, wid = threadIdx.x >> 5;
  const int waveM = wid >> 1, waveN = wid & 1;
  const int n = lane & 15, hh = lane >> 4;
#pragma unroll
  for (int i = 0; i < 2; ++i)
#pragma unroll
    for (int j = 0; j < 2; ++j) {
      const int col = colBase + waveN * 32 + j * 16 + n;
      const float bv = bias[col];
#pragma unroll
      for (int v = 0; v < 8; ++v) {
        const int row = rowBase + waveM * 32 + i * 16 + (hh << 3) + v;
        out[(size_t)row * DMODEL + col] = acc[i][j][v] + bv;
      }
    }
}

extern "C" void kernel_launch(void* const* d_in, const int* in_sizes, int n_in,
                              void* d_out, int out_size, void* d_ws, size_t ws_size,
                              hipStream_t stream) {
  (void)in_sizes; (void)n_in; (void)out_size; (void)ws_size;
  const float* x    = (const float*)d_in[0];
  const float* Wqkv = (const float*)d_in[1];
  const float* bqkv = (const float*)d_in[2];
  const float* Wout = (const float*)d_in[3];
  const float* bout = (const float*)d_in[4];
  float* out = (float*)d_out;

  // carve f16 scratch from d_ws (256B aligned)
  char* ws = (char*)d_ws;
  size_t off = 0;
  auto carve = [&](size_t halves) {
    void* p = ws + off;
    off = (off + halves * sizeof(_Float16) + 255) & ~(size_t)255;
    return (_Float16*)p;
  };
  _Float16* xb    = carve((size_t)MROWS * DMODEL);
  _Float16* wqkvb = carve((size_t)DMODEL * NQKV);
  _Float16* woutb = carve((size_t)DMODEL * DMODEL);
  _Float16* qb    = carve((size_t)NBATCH * NHEAD * NTOK * HDIM);
  _Float16* kb    = carve((size_t)NBATCH * NHEAD * NTOK * HDIM);
  _Float16* vb    = carve((size_t)NBATCH * NHEAD * NTOK * HDIM);
  _Float16* attnb = carve((size_t)MROWS * DMODEL);

  // 1) f32 -> f16 conversions (vectorized)
  {
    int n4;
    n4 = MROWS * DMODEL / 4;
    cvt_f32_to_f16<<<(n4 + 255) / 256, 256, 0, stream>>>((const float4*)x, xb, n4);
    n4 = DMODEL * NQKV / 4;
    cvt_f32_to_f16<<<(n4 + 255) / 256, 256, 0, stream>>>((const float4*)Wqkv, wqkvb, n4);
    n4 = DMODEL * DMODEL / 4;
    cvt_f32_to_f16<<<(n4 + 255) / 256, 256, 0, stream>>>((const float4*)Wout, woutb, n4);
  }
  // 2) QKV projection (WMMA + double-buffered async-LDS staging)
  gemm_qkv_kernel<<<dim3(NQKV / 64, MROWS / 128), 256, 0, stream>>>(xb, wqkvb, bqkv, qb, kb, vb);
  // 3) Flash attention (WMMA + online softmax + double-buffered async K/V)
  attn_kernel<<<dim3(NTOK / 64, NBATCH * NHEAD), 128, 0, stream>>>(qb, kb, vb, attnb);
  // 4) Output projection (WMMA) -> f32 + bias
  gemm_out_kernel<<<dim3(DMODEL / 64, MROWS / 128), 256, 0, stream>>>(attnb, woutb, bout, out);
}
